// RNAPocketEncoder_25022479466500
// MI455X (gfx1250) — compile-verified, matrix operand order checked by reading, not run on hardware
//
#include <hip/hip_runtime.h>
#include <hip/hip_bf16.h>

#define N_NODES 4096
#define N_EDGES 32768
#define N_BATCH 16
#define N_LAYERS 4
#define WDIM 3456
#define NT 216            // WDIM / 16
#define HROW 120          // 32 (0e) + 48 (1o) + 40 (2e)
#define F2DIM 592         // expanded left-factor table per edge

typedef __bf16 v16bf __attribute__((ext_vector_type(16)));
typedef float  v8f   __attribute__((ext_vector_type(8)));
typedef float  f32x4 __attribute__((ext_vector_type(4)));
typedef unsigned int u32x4 __attribute__((ext_vector_type(4)));

// ---------- constants ----------
#define SQ3f  1.73205080757f
#define SQ5f  2.23606797750f
#define SQ15f 3.87298334621f
#define A0f   0.13363062095f   // sqrt(1/56)
#define A1f   0.20412414523f   // sqrt(3/72)
#define A2f   0.27950849718f   // sqrt(5/64)
#define A0S3f 0.07715167498f   // A0/sqrt3
#define A0S5f 0.05976143047f   // A0/sqrt5
#define A1S3f 0.11785113020f   // A1/sqrt3
// C112 coefficients (Q[k][i][j]/sqrt(7.5))
#define C1f 0.31622776601f
#define C2f 0.18257418584f
#define C3f 0.36514837167f
// C222 coefficients (T/||T||, ||T||=sqrt(315/32))
#define D1f 0.23904572186f
#define D2f 0.11952286093f
#define D3f 0.20701966780f

__device__ inline unsigned short f2bf(float f) {
  union { float f; unsigned u; } x; x.f = f;
  unsigned u = x.u;
  return (unsigned short)((u + 0x7FFFu + ((u >> 16) & 1u)) >> 16);
}
__device__ inline float silu(float v) { return v * (1.f / (1.f + __expf(-v))); }

// ---------- CG contraction helpers (sparse, hardcoded) ----------
__device__ inline float t121f(const float* a, const float* g, int j) {
  if (j == 0) return a[0]*(-C2f*g[2]-C1f*g[4]) + a[1]*(C1f*g[1]) + a[2]*(C1f*g[0]);
  if (j == 1) return a[0]*(C1f*g[1]) + a[1]*(C3f*g[2]) + a[2]*(C1f*g[3]);
  return          a[0]*(C1f*g[0]) + a[1]*(C1f*g[3]) + a[2]*(-C2f*g[2]+C1f*g[4]);
}
__device__ inline float t211f(const float* w, const float* s1, int c) {
  if (c == 0) return s1[0]*(-C2f*w[2]-C1f*w[4]) + s1[1]*(C1f*w[1]) + s1[2]*(C1f*w[0]);
  if (c == 1) return s1[0]*(C1f*w[1]) + s1[1]*(C3f*w[2]) + s1[2]*(C1f*w[3]);
  return          s1[0]*(C1f*w[0]) + s1[1]*(C1f*w[3]) + s1[2]*(-C2f*w[2]+C1f*w[4]);
}
__device__ inline float t112f(const float* a, const float* b, int k) {
  if (k == 0) return C1f*(a[0]*b[2] + a[2]*b[0]);
  if (k == 1) return C1f*(a[0]*b[1] + a[1]*b[0]);
  if (k == 2) return -C2f*a[0]*b[0] + C3f*a[1]*b[1] - C2f*a[2]*b[2];
  if (k == 3) return C1f*(a[1]*b[2] + a[2]*b[1]);
  return          C1f*(a[2]*b[2] - a[0]*b[0]);
}
__device__ inline float t222f(const float* w, const float* g, int k) {
  if (k == 0) return -D1f*(w[0]*g[2]+w[2]*g[0]) + D3f*(w[1]*g[3]+w[3]*g[1]);
  if (k == 1) return  D2f*(w[1]*g[2]+w[2]*g[1]) - D3f*(w[1]*g[4]+w[4]*g[1]) + D3f*(w[0]*g[3]+w[3]*g[0]);
  if (k == 2) return -D1f*w[0]*g[0] + D2f*w[1]*g[1] + D1f*w[2]*g[2] + D2f*w[3]*g[3] - D1f*w[4]*g[4];
  if (k == 3) return  D2f*(w[3]*g[2]+w[2]*g[3]) + D3f*(w[3]*g[4]+w[4]*g[3]) + D3f*(w[0]*g[1]+w[1]*g[0]);
  return          -D1f*(w[4]*g[2]+w[2]*g[4]) - D3f*w[1]*g[1] + D3f*w[3]*g[3];
}

// expanded left-factor table, 592 slots per edge, s1/s2 folded into 011/022
// layout: [0)F000(32) [32)F110(16) [48)F220(8) [56)F011v(96) [152)F101(48)
//         [200)F121(48) [248)F211(24) [272)F022v(160) [432)F112(80)
//         [512)F202(40) [552)F222(40)
__device__ inline float computeF2(int s, const float* hs, const float* g) {
  const float* s1 = g;
  const float* s2 = g + 3;
  if (s < 32)  return A0f * hs[s];
  if (s < 48)  { const float* a = hs + 32 + (s-32)*3;
                 return A0S3f*(a[0]*s1[0] + a[1]*s1[1] + a[2]*s1[2]); }
  if (s < 56)  { const float* w = hs + 80 + (s-48)*5;
                 float v = 0.f;
                 #pragma unroll
                 for (int k = 0; k < 5; ++k) v += w[k]*s2[k];
                 return A0S5f * v; }
  if (s < 152) { int idx = s-56;  return A1S3f * hs[idx/3] * s1[idx%3]; }
  if (s < 200) { return A1S3f * hs[32 + (s-152)]; }
  if (s < 248) { int idx = s-200; return A1f * t121f(hs+32+(idx/3)*3, s2, idx%3); }
  if (s < 272) { int idx = s-248; return A1f * t211f(hs+80+(idx/3)*5, s1, idx%3); }
  if (s < 432) { int idx = s-272; return 0.125f * hs[idx/5] * s2[idx%5]; }
  if (s < 512) { int idx = s-432; return A2f * t112f(hs+32+(idx/5)*3, s1, idx%5); }
  if (s < 552) { return 0.125f * hs[80 + (s-512)]; }
  { int idx = s-552; return A2f * t222f(hs+80+(idx/5)*5, s2, idx%5); }
}

// ---------- compile-time-specialized WMMA + TP-consume loop for one path ----------
template<int CSTART, int FOFF, int NC, int NTBEG, int NTEND>
__device__ __forceinline__ void tp_path(int wave, int lane, int ebase, int layer,
                                        const unsigned short* __restrict__ W3p,
                                        const float* __restrict__ b3,
                                        v16bf a0v, v16bf a1v,
                                        const float (*sF2T)[16], float (*sMT)[16]) {
  constexpr int T2    = NC >> 1;              // 0,1,2
  constexpr int SHIFT = 5 - T2;               // 5,4,3
  constexpr int VMASK = (32 >> T2) - 1;       // 31,15,7
  constexpr int MOFF  = T2*32 + (T2 >> 1)*16; // 0,32,80
  union BF { v16bf v; u32x4 q[2]; };
  for (int nt = NTBEG + wave; nt < NTEND; nt += 4) {
    size_t boff = (((size_t)layer*NT + nt) * 2) * 512 + (size_t)lane * 16;
    const u32x4* p0 = (const u32x4*)(W3p + boff);
    const u32x4* p1 = (const u32x4*)(W3p + boff + 512);
    BF b0, b1;
    b0.q[0] = p0[0]; b0.q[1] = p0[1];
    b1.q[0] = p1[0]; b1.q[1] = p1[1];
    __builtin_prefetch(W3p + boff + 4*1024, 0, 1);

    int col0   = nt*16 + (lane & 15);
    float bias = b3[col0];
    v8f acc = {};
    acc = __builtin_amdgcn_wmma_f32_16x16x32_bf16(false, a0v, false, b0.v, (short)0, acc, false, false);
    acc = __builtin_amdgcn_wmma_f32_16x16x32_bf16(false, a1v, false, b1.v, (short)0, acc, false, false);

    int cc = col0 - CSTART;
    int u  = cc >> SHIFT;
    int v  = cc & VMASK;
    int f  = FOFF + u*NC;
    int mb = MOFF + v*NC;
    float wv[8];
    #pragma unroll
    for (int r = 0; r < 8; ++r) wv[r] = acc[r] + bias;
    #pragma unroll
    for (int j = 0; j < NC; ++j) {
      const float* Fr = &sF2T[f + j][ebase];
      float*       Mr = &sMT[mb + j][ebase];
      f32x4 fa = *(const f32x4*)(Fr);
      f32x4 fb = *(const f32x4*)(Fr + 4);
      atomicAdd(&Mr[0], wv[0]*fa[0]);
      atomicAdd(&Mr[1], wv[1]*fa[1]);
      atomicAdd(&Mr[2], wv[2]*fa[2]);
      atomicAdd(&Mr[3], wv[3]*fa[3]);
      atomicAdd(&Mr[4], wv[4]*fb[0]);
      atomicAdd(&Mr[5], wv[5]*fb[1]);
      atomicAdd(&Mr[6], wv[6]*fb[2]);
      atomicAdd(&Mr[7], wv[7]*fb[3]);
    }
  }
}

// ---------- kernel: repack rad_W3 into WMMA bf16 B-fragment layout ----------
__global__ void k_pack_w3(const float* __restrict__ W3, unsigned short* __restrict__ P) {
  const int PER_L = NT * 2 * 32 * 16;    // 221184
  int idx = blockIdx.x * 256 + threadIdx.x;
  if (idx >= N_LAYERS * PER_L) return;
  int l = idx / PER_L;   int r  = idx % PER_L;
  int nt = r / 1024;     int r2 = r % 1024;
  int kc = r2 >> 9;      int r3 = r2 & 511;
  int lane = r3 >> 4;    int q  = r3 & 15;
  int col  = nt * 16 + (lane & 15);
  int koff = (lane < 16) ? 0 : 8;
  int k    = kc * 32 + koff + (q < 8 ? q : q + 8);
  P[idx] = f2bf(W3[((size_t)l * 64 + k) * WDIM + col]);
}

// ---------- kernel: per-edge geometry (s1, s2, rbf) ----------
__global__ void k_geom(const float* __restrict__ pos, const int* __restrict__ ei,
                       float* __restrict__ geom) {
  int e = blockIdx.x * 256 + threadIdx.x;
  if (e >= N_EDGES) return;
  int s = ei[e], d = ei[N_EDGES + e];
  float rx = pos[d*3+0] - pos[s*3+0];
  float ry = pos[d*3+1] - pos[s*3+1];
  float rz = pos[d*3+2] - pos[s*3+2];
  float dist = fmaxf(sqrtf(rx*rx + ry*ry + rz*rz), 1e-6f);
  float x = rx/dist, y = ry/dist, z = rz/dist;
  float* g = geom + (size_t)e * 16;
  g[0] = SQ3f*x;  g[1] = SQ3f*y;  g[2] = SQ3f*z;
  g[3] = SQ15f*x*z;
  g[4] = SQ15f*x*y;
  g[5] = SQ5f*(y*y - 0.5f*(x*x + z*z));
  g[6] = SQ15f*y*z;
  g[7] = 0.5f*SQ15f*(z*z - x*x);
  #pragma unroll
  for (int c = 0; c < 8; ++c) {
    float dd = dist - (float)c * (6.0f / 7.0f);
    g[8 + c] = 0.79788456f * __expf(-2.0f * dd * dd);
  }
}

// ---------- kernel: input embedding ----------
__global__ void k_embed(const float* __restrict__ x, const float* __restrict__ embW,
                        float* __restrict__ h) {
  int t = blockIdx.x * 256 + threadIdx.x;
  if (t >= N_NODES * HROW) return;
  int n = t / HROW, s = t % HROW;
  float v = 0.f;
  if (s < 32) {
    for (int d = 0; d < 64; ++d) v += x[n*64 + d] * embW[d*32 + s];
    v *= 0.125f;
  }
  h[t] = v;
}

// ---------- kernel: fused radial-MLP (WMMA) + tensor-product messages ----------
__launch_bounds__(128)
__global__ void k_messages(int layer, const float* __restrict__ h, const float* __restrict__ geom,
                           float* __restrict__ agg, const int* __restrict__ ei,
                           const unsigned short* __restrict__ W3p,
                           const float* __restrict__ rW1, const float* __restrict__ rb1,
                           const float* __restrict__ rW2, const float* __restrict__ rb2,
                           const float* __restrict__ rb3) {
  __shared__ int   sSrc[16], sDst[16];
  __shared__ float sG[16][16];
  __shared__ float sHS[16][HROW];
  __shared__ float sT1[16][64];
  __shared__ unsigned short sHB[16][64];
  __shared__ float sF2T[F2DIM][16];   // transposed: factor-major, edge-minor
  __shared__ float sMT[HROW][16];     // transposed: slot-major, edge-minor

  const int tid = threadIdx.x;
  const int e0  = blockIdx.x * 16;

  // stage 1a: edge indices + geometry
  if (tid < 16) { sSrc[tid] = ei[e0 + tid]; sDst[tid] = ei[N_EDGES + e0 + tid]; }
  {
    int i = tid;       sG[i >> 4][i & 15] = geom[(size_t)e0*16 + i];
    i += 128;          sG[i >> 4][i & 15] = geom[(size_t)e0*16 + i];
  }
  __syncthreads();

  // stage 1b: gather source node features + radial layer 1
  for (int i = tid; i < 16 * HROW; i += 128) {
    int e = i / HROW, s = i % HROW;
    sHS[e][s] = h[(size_t)sSrc[e]*HROW + s];
  }
  {
    int e = tid & 15, part = tid >> 4;
    const float* W1 = rW1 + (size_t)layer*8*64;
    const float* b1 = rb1 + (size_t)layer*64;
    #pragma unroll
    for (int oo = 0; oo < 8; ++oo) {
      int o = part*8 + oo;
      float v = b1[o];
      #pragma unroll
      for (int i2 = 0; i2 < 8; ++i2) v += sG[e][8 + i2] * W1[i2*64 + o];
      sT1[e][o] = silu(v);
    }
  }
  __syncthreads();

  // stage 2: radial layer 2 -> bf16 activations (A matrix for WMMA)
  {
    int e = tid & 15, part = tid >> 4;
    const float* W2 = rW2 + (size_t)layer*64*64 + part*8;
    const float* b2 = rb2 + (size_t)layer*64 + part*8;
    float acc8[8];
    #pragma unroll
    for (int oo = 0; oo < 8; ++oo) acc8[oo] = b2[oo];
    for (int i2 = 0; i2 < 64; ++i2) {
      float t = sT1[e][i2];
      f32x4 wA = *(const f32x4*)(W2 + i2*64);
      f32x4 wB = *(const f32x4*)(W2 + i2*64 + 4);
      acc8[0] += t*wA[0]; acc8[1] += t*wA[1]; acc8[2] += t*wA[2]; acc8[3] += t*wA[3];
      acc8[4] += t*wB[0]; acc8[5] += t*wB[1]; acc8[6] += t*wB[2]; acc8[7] += t*wB[3];
    }
    #pragma unroll
    for (int oo = 0; oo < 8; ++oo) sHB[e][part*8 + oo] = f2bf(silu(acc8[oo]));
  }
  __syncthreads();

  // stage 3: expanded left factors (transposed) + zero message accumulator
  for (int i = tid; i < 16 * HROW; i += 128) sMT[i >> 4][i & 15] = 0.f;
  for (int i = tid; i < 16 * F2DIM; i += 128) {
    int e = i & 15, s = i >> 4;   // wave shares slot index -> mostly uniform branches
    sF2T[s][e] = computeF2(s, sHS[e], sG[e]);
  }
  __syncthreads();

  // stage 4: WMMA GEMM over 216 column tiles, compile-time path specialization
  const int lane  = tid & 31;
  const int wave  = tid >> 5;
  const int ebase = (lane < 16) ? 0 : 8;
  union AF { v16bf v; unsigned short s[16]; };
  AF a0, a1;
  {
    int m = lane & 15;
    int koff = (lane < 16) ? 0 : 8;
    #pragma unroll
    for (int q = 0; q < 16; ++q) {
      int k = koff + (q < 8 ? q : q + 8);
      a0.s[q] = sHB[m][k];
      a1.s[q] = sHB[m][32 + k];
    }
  }
  const float* b3 = rb3 + (size_t)layer * WDIM;
  // per-path tile counts (16,8,4,8,4,4,2,4,2,1,1 per wave) all divide evenly across 4 waves
  tp_path<0,    0,   1, 0,   64 >(wave, lane, ebase, layer, W3p, b3, a0.v, a1.v, sF2T, sMT);
  tp_path<1024, 32,  1, 64,  96 >(wave, lane, ebase, layer, W3p, b3, a0.v, a1.v, sF2T, sMT);
  tp_path<1536, 48,  1, 96,  112>(wave, lane, ebase, layer, W3p, b3, a0.v, a1.v, sF2T, sMT);
  tp_path<1792, 56,  3, 112, 144>(wave, lane, ebase, layer, W3p, b3, a0.v, a1.v, sF2T, sMT);
  tp_path<2304, 152, 3, 144, 160>(wave, lane, ebase, layer, W3p, b3, a0.v, a1.v, sF2T, sMT);
  tp_path<2560, 200, 3, 160, 176>(wave, lane, ebase, layer, W3p, b3, a0.v, a1.v, sF2T, sMT);
  tp_path<2816, 248, 3, 176, 184>(wave, lane, ebase, layer, W3p, b3, a0.v, a1.v, sF2T, sMT);
  tp_path<2944, 272, 5, 184, 200>(wave, lane, ebase, layer, W3p, b3, a0.v, a1.v, sF2T, sMT);
  tp_path<3200, 432, 5, 200, 208>(wave, lane, ebase, layer, W3p, b3, a0.v, a1.v, sF2T, sMT);
  tp_path<3328, 512, 5, 208, 212>(wave, lane, ebase, layer, W3p, b3, a0.v, a1.v, sF2T, sMT);
  tp_path<3392, 552, 5, 212, 216>(wave, lane, ebase, layer, W3p, b3, a0.v, a1.v, sF2T, sMT);
  __syncthreads();

  // stage 5: scatter messages to destination nodes
  for (int i = tid; i < 16 * HROW; i += 128) {
    int e = i & 15, s = i >> 4;
    atomicAdd(&agg[(size_t)sDst[e]*HROW + s], sMT[s][e]);
  }
}

// ---------- kernel: self-interaction + aggregated messages ----------
__launch_bounds__(128)
__global__ void k_update(int layer, float* __restrict__ h, const float* __restrict__ agg,
                         const float* __restrict__ siW0, const float* __restrict__ siW1,
                         const float* __restrict__ siW2) {
  __shared__ float hold[HROW];
  int n = blockIdx.x, t = threadIdx.x;
  if (t < HROW) hold[t] = h[(size_t)n*HROW + t];
  __syncthreads();
  if (t >= HROW) return;
  float v;
  if (t < 32) {
    const float* W = siW0 + (size_t)layer*32*32;
    float s = 0.f;
    for (int u = 0; u < 32; ++u) s += hold[u] * W[u*32 + t];
    v = 0.17677670f * s;
  } else if (t < 80) {
    int idx = t - 32, vv = idx / 3, i = idx % 3;
    const float* W = siW1 + (size_t)layer*16*16;
    float s = 0.f;
    for (int u = 0; u < 16; ++u) s += hold[32 + u*3 + i] * W[u*16 + vv];
    v = 0.25f * s;
  } else {
    int idx = t - 80, vv = idx / 5, k = idx % 5;
    const float* W = siW2 + (size_t)layer*8*8;
    float s = 0.f;
    for (int u = 0; u < 8; ++u) s += hold[80 + u*5 + k] * W[u*8 + vv];
    v = 0.35355339f * s;
  }
  h[(size_t)n*HROW + t] = v + agg[(size_t)n*HROW + t];
}

// ---------- kernel: attention-logit MLP ----------
__launch_bounds__(128)
__global__ void k_att(const float* __restrict__ h,
                      const float* __restrict__ W1, const float* __restrict__ b1,
                      const float* __restrict__ W2, const float* __restrict__ b2,
                      const float* __restrict__ W3, const float* __restrict__ b3,
                      float* __restrict__ a) {
  __shared__ float h0[32], a1[128], red[128];
  int n = blockIdx.x, t = threadIdx.x;
  if (t < 32) h0[t] = h[(size_t)n*HROW + t];
  __syncthreads();
  float v = b1[t];
  for (int u = 0; u < 32; ++u) v += h0[u] * W1[u*128 + t];
  a1[t] = silu(v);
  __syncthreads();
  float v2 = b2[t];
  for (int i = 0; i < 128; ++i) v2 += a1[i] * W2[i*128 + t];
  red[t] = silu(v2) * W3[t];
  __syncthreads();
  for (int o = 64; o > 0; o >>= 1) {
    if (t < o) red[t] += red[t + o];
    __syncthreads();
  }
  if (t == 0) a[n] = red[0] + b3[0];
}

// ---------- kernel: segment softmax pooling + output head + LayerNorm ----------
__launch_bounds__(256)
__global__ void k_pool(const float* __restrict__ h, const float* __restrict__ a,
                       const int* __restrict__ batch,
                       const float* __restrict__ outW, const float* __restrict__ outB,
                       const float* __restrict__ lng, const float* __restrict__ lnb,
                       float* __restrict__ out) {
  __shared__ float red[256];
  __shared__ float gsh[32];
  __shared__ float osh[512];
  __shared__ float amax_s, den_s, mu_s, rs_s;
  int b = blockIdx.x, t = threadIdx.x;

  float m = -3.0e38f;
  for (int n = t; n < N_NODES; n += 256) if (batch[n] == b) m = fmaxf(m, a[n]);
  red[t] = m; __syncthreads();
  for (int o = 128; o > 0; o >>= 1) { if (t < o) red[t] = fmaxf(red[t], red[t+o]); __syncthreads(); }
  if (t == 0) amax_s = red[0];
  __syncthreads();
  float amax = amax_s;

  float sE = 0.f;
  for (int n = t; n < N_NODES; n += 256) if (batch[n] == b) sE += __expf(a[n] - amax);
  red[t] = sE; __syncthreads();
  for (int o = 128; o > 0; o >>= 1) { if (t < o) red[t] += red[t+o]; __syncthreads(); }
  if (t == 0) den_s = red[0];
  if (t < 32) gsh[t] = 0.f;
  __syncthreads();
  float den = den_s;

  {
    int v = t & 31, grp = t >> 5;
    float p = 0.f;
    for (int n = grp; n < N_NODES; n += 8) if (batch[n] == b) {
      float att = __expf(a[n] - amax) / den;
      p += att * h[(size_t)n*HROW + v];
    }
    atomicAdd(&gsh[v], p);
  }
  __syncthreads();

  for (int j = t; j < 512; j += 256) {
    float v = outB[j];
    for (int u = 0; u < 32; ++u) v += gsh[u] * outW[u*512 + j];
    osh[j] = v;
  }
  __syncthreads();

  float s1 = 0.f, s2 = 0.f;
  for (int j = t; j < 512; j += 256) { s1 += osh[j]; s2 += osh[j]*osh[j]; }
  red[t] = s1; __syncthreads();
  for (int o = 128; o > 0; o >>= 1) { if (t < o) red[t] += red[t+o]; __syncthreads(); }
  if (t == 0) mu_s = red[0] / 512.f;
  __syncthreads();
  red[t] = s2; __syncthreads();
  for (int o = 128; o > 0; o >>= 1) { if (t < o) red[t] += red[t+o]; __syncthreads(); }
  if (t == 0) {
    float var = red[0] / 512.f - mu_s * mu_s;
    rs_s = rsqrtf(var + 1e-5f);
  }
  __syncthreads();
  for (int j = t; j < 512; j += 256)
    out[(size_t)b*512 + j] = (osh[j] - mu_s) * rs_s * lng[j] + lnb[j];
}

// ---------- launcher ----------
extern "C" void kernel_launch(void* const* d_in, const int* in_sizes, int n_in,
                              void* d_out, int out_size, void* d_ws, size_t ws_size,
                              hipStream_t stream) {
  const float* x    = (const float*)d_in[0];
  const float* pos  = (const float*)d_in[1];
  const float* embW = (const float*)d_in[2];
  const float* siW0 = (const float*)d_in[3];
  const float* siW1 = (const float*)d_in[4];
  const float* siW2 = (const float*)d_in[5];
  const float* rW1  = (const float*)d_in[6];
  const float* rb1  = (const float*)d_in[7];
  const float* rW2  = (const float*)d_in[8];
  const float* rb2  = (const float*)d_in[9];
  const float* rW3  = (const float*)d_in[10];
  const float* rb3  = (const float*)d_in[11];
  const float* pW1  = (const float*)d_in[12];
  const float* pb1  = (const float*)d_in[13];
  const float* pW2  = (const float*)d_in[14];
  const float* pb2  = (const float*)d_in[15];
  const float* pW3  = (const float*)d_in[16];
  const float* pb3  = (const float*)d_in[17];
  const float* outW = (const float*)d_in[18];
  const float* outB = (const float*)d_in[19];
  const float* lng  = (const float*)d_in[20];
  const float* lnb  = (const float*)d_in[21];
  const int*   ei   = (const int*)d_in[22];
  const int*   bat  = (const int*)d_in[23];

  char* w = (char*)d_ws;
  size_t o = 0;
  unsigned short* W3p = (unsigned short*)(w + o); o += (size_t)N_LAYERS * NT * 2 * 32 * 16 * 2; // 1.77 MB
  float* geom = (float*)(w + o); o += (size_t)N_EDGES * 16 * 4;    // 2.10 MB
  float* h    = (float*)(w + o); o += (size_t)N_NODES * HROW * 4;  // 1.97 MB
  float* agg  = (float*)(w + o); o += (size_t)N_NODES * HROW * 4;  // 1.97 MB
  float* aatt = (float*)(w + o); o += (size_t)N_NODES * 4;

  k_pack_w3<<<(N_LAYERS * NT * 2 * 32 * 16 + 255) / 256, 256, 0, stream>>>(rW3, W3p);
  k_geom<<<(N_EDGES + 255) / 256, 256, 0, stream>>>(pos, ei, geom);
  k_embed<<<(N_NODES * HROW + 255) / 256, 256, 0, stream>>>(x, embW, h);

  for (int l = 0; l < N_LAYERS; ++l) {
    hipMemsetAsync(agg, 0, (size_t)N_NODES * HROW * 4, stream);
    k_messages<<<N_EDGES / 16, 128, 0, stream>>>(l, h, geom, agg, ei, W3p,
                                                 rW1, rb1, rW2, rb2, rb3);
    k_update<<<N_NODES, 128, 0, stream>>>(l, h, agg, siW0, siW1, siW2);
  }

  k_att<<<N_NODES, 128, 0, stream>>>(h, pW1, pb1, pW2, pb2, pW3, pb3, aatt);
  k_pool<<<N_BATCH, 256, 0, stream>>>(h, aatt, bat, outW, outB, lng, lnb, (float*)d_out);
}
